// SimpleLSTM_17360257810697
// MI455X (gfx1250) — matrix operand order repaired and checked by reading, not compiled, not run on hardware
//
#include <hip/hip_runtime.h>

// ---------------------------------------------------------------------------
// LSTM (B=2048, T=512, I=1, H=64, O=1) on CDNA5 / gfx1250, wave32 WMMA.
//
// 4 waves (one per SIMD of a WGP) cooperate on one 16-row batch tile.
// Wave n owns hidden chunk n (cols 16n..16n+15 of h/c): 8 x
// v_wmma_f32_16x16x32_bf16 per step. W_hh bf16 B-tiles register-resident.
// c stays in f32 registers. h is exchanged via a ping-pong LDS bounce into
// the 16-bit A-matrix layout -> ONE workgroup barrier per step.
// Activations use native V_TANH_F32. FC head (O=1) fused via shfl_xor +
// LDS partial combine. x chunks are prefetched with CDNA5
// GLOBAL_LOAD_ASYNC_TO_LDS_B32 (ASYNCcnt) into a double buffer, overlapped
// with compute.
// ---------------------------------------------------------------------------

typedef __attribute__((ext_vector_type(16))) __bf16 v16bf;
typedef __attribute__((ext_vector_type(8)))  float  v8f;

#define TSEQ   512
#define BATCH  2048
#define HDIM   64
#define GDIM   256   // 4*H
#define XCHUNK 64    // timesteps of x staged in LDS per buffer

union V16 { v16bf v; float4 f[2]; };

__device__ __forceinline__ unsigned short f2bf(float x) {
  unsigned u = __builtin_bit_cast(unsigned, x);
  u += 0x7FFFu + ((u >> 16) & 1u);          // round-to-nearest-even
  return (unsigned short)(u >> 16);
}

#if __has_builtin(__builtin_amdgcn_tanhf)
__device__ __forceinline__ float fast_tanh(float x) {
  return __builtin_amdgcn_tanhf(x);         // v_tanh_f32 (CDNA5 trans op)
}
#else
__device__ __forceinline__ float fast_tanh(float x) {
  float r;
  asm("v_tanh_f32 %0, %1" : "=v"(r) : "v"(x));
  return r;
}
#endif

__device__ __forceinline__ float fast_sigmoid(float x) {
  // sigmoid(x) = 0.5*tanh(x/2) + 0.5  -> 1 trans + 2 VALU
  return __builtin_fmaf(fast_tanh(0.5f * x), 0.5f, 0.5f);
}

// LDS byte offset of a __shared__ object: low 32 bits of the generic address
// (flat->LDS aperture mapping truncates to addr[31:0]).
__device__ __forceinline__ unsigned lds_off(const void* p) {
  return (unsigned)(unsigned long long)p;
}

// Async global -> LDS copy of one x chunk (transposed to [t][row]).
// 8 x GLOBAL_LOAD_ASYNC_TO_LDS_B32 per lane, tracked by ASYNCcnt.
__device__ __forceinline__ void prefetch_x_async(const float* __restrict__ x,
                                                 int base, int tc,
                                                 float* dstbuf, int tid) {
#pragma unroll
  for (int k = 0; k < 8; ++k) {
    int i  = tid + k * 128;
    int r  = i >> 6;              // XCHUNK == 64
    int tt = i & (XCHUNK - 1);
    unsigned     lds = lds_off(&dstbuf[tt * 16 + r]);
    const float* g   = &x[(base + r) * TSEQ + tc + tt];
    asm volatile("global_load_async_to_lds_b32 %0, %1, off"
                 :: "v"(lds), "v"(g) : "memory");
  }
}

__device__ __forceinline__ void wait_async_zero() {
  asm volatile("s_wait_asynccnt 0x0" ::: "memory");
}

__global__ __launch_bounds__(128)
void lstm_wmma_kernel(const float* __restrict__ x,
                      const float* __restrict__ w_ih,
                      const float* __restrict__ w_hh,
                      const float* __restrict__ b_ih,
                      const float* __restrict__ b_hh,
                      const float* __restrict__ w_fc,
                      const float* __restrict__ b_fc,
                      float* __restrict__ out) {
  __shared__ unsigned short s_whh[GDIM * HDIM];     // 32 KB bf16 [gate][hid]
  __shared__ float          s_x[2][XCHUNK * 16];    // 2 x 4 KB, [t][row]
  __shared__ unsigned short s_h[2][16 * HDIM];      // 2 x 2 KB bf16 [m][k]
  __shared__ float          s_fc[2][4 * 16];        // fc partials [wave][row]

  const int tid  = threadIdx.x;       // 0..127
  const int wv   = tid >> 5;          // wave id == hidden chunk n (0..3)
  const int lane = tid & 31;          // 0..31 within wave
  const int base = blockIdx.x * 16;   // batch tile base
  const int lm   = lane & 15;         // N column / M row within half
  const int lh   = lane >> 4;         // lane half (0/1)

  // ---- one-time: W_hh -> bf16 LDS (coalesced, layout already [N][K]) ----
  for (int i = tid; i < GDIM * HDIM; i += 128) s_whh[i] = f2bf(w_hh[i]);

  // ---- async prefetch of the first x chunk ----
  prefetch_x_async(x, base, 0, s_x[0], tid);

  // ---- per-lane scalar params (this wave's 4 gate tiles) ----
  float wihr[4], biasr[4];
#pragma unroll
  for (int g = 0; g < 4; ++g) {
    int gate = g * 64 + wv * 16 + lm;
    wihr[g]  = w_ih[gate];
    biasr[g] = b_ih[gate] + b_hh[gate];
  }
  const float wfcl = w_fc[wv * 16 + lm];
  const float bfc  = b_fc[0];

  wait_async_zero();
  __syncthreads();   // s_whh + s_x[0] ready for all waves

  // ---- this wave's W_hh B-tiles -> registers (this code's contiguous k order: lane=N,
  //      K packed 2/VGPR, lanes 16..31 hold K+16): two ds_load_b128/tile ----
  V16 barr[4][2];
#pragma unroll
  for (int g = 0; g < 4; ++g) {
#pragma unroll
    for (int kh = 0; kh < 2; ++kh) {
      const unsigned short* p =
          &s_whh[(g * 64 + wv * 16 + lm) * HDIM + kh * 32 + lh * 16];
      __builtin_memcpy(&barr[g][kh].f[0], p,     16);
      __builtin_memcpy(&barr[g][kh].f[1], p + 8, 16);
    }
  }

  // ---- state: full h (A-layout bf16, two K-halves), own c chunk (f32) ----
  V16 a0, a1;
  a0.f[0] = make_float4(0.f, 0.f, 0.f, 0.f);
  a0.f[1] = a0.f[0];
  a1 = a0;
  v8f cst = {};   // c[16 x 16] chunk wv, D-layout

  int pb = 0;     // x chunk buffer parity
  int pp = 0;     // h / fc ping-pong parity

#pragma unroll 1
  for (int tc = 0; tc < TSEQ; tc += XCHUNK) {
    // overlap: async prefetch of the next chunk into the other buffer
    if (tc + XCHUNK < TSEQ)
      prefetch_x_async(x, base, tc + XCHUNK, s_x[pb ^ 1], tid);

#pragma unroll 1
    for (int tt = 0; tt < XCHUNK; ++tt) {
      const int t = tc + tt;

      // 8 per-row x values for this lane (D-layout rows m = r + 8*lh)
      float4 xa, xb;
      const float* px = &s_x[pb][tt * 16 + lh * 8];
      __builtin_memcpy(&xa, px,     16);
      __builtin_memcpy(&xb, px + 4, 16);
      const float xr[8] = {xa.x, xa.y, xa.z, xa.w, xb.x, xb.y, xb.z, xb.w};

      // ---- gates for this hidden chunk: i,f,g,o tiles ----
      v8f acc[4];
#pragma unroll
      for (int g = 0; g < 4; ++g) {
#pragma unroll
        for (int r = 0; r < 8; ++r)
          acc[g][r] = __builtin_fmaf(xr[r], wihr[g], biasr[g]);
        acc[g] = __builtin_amdgcn_wmma_f32_16x16x32_bf16(
            false, a0.v, false, barr[g][0].v, (short)0, acc[g], false, false);
        acc[g] = __builtin_amdgcn_wmma_f32_16x16x32_bf16(
            false, a1.v, false, barr[g][1].v, (short)0, acc[g], false, false);
      }

      // ---- activations + state update + h staging + fc partial ----
      float fcp[8];
#pragma unroll
      for (int r = 0; r < 8; ++r) {
        float iv = fast_sigmoid(acc[0][r]);
        float fv = fast_sigmoid(acc[1][r]);
        float gv = fast_tanh(acc[2][r]);
        float ov = fast_sigmoid(acc[3][r]);
        float cv = __builtin_fmaf(fv, cst[r], iv * gv);
        cst[r] = cv;
        float hv = ov * fast_tanh(cv);
        fcp[r] = hv * wfcl;
        s_h[pp][(r + lh * 8) * HDIM + wv * 16 + lm] = f2bf(hv);
      }

      // fc: butterfly-reduce this chunk's partial over 16 lanes per half
#pragma unroll
      for (int r = 0; r < 8; ++r) {
#pragma unroll
        for (int m = 8; m >= 1; m >>= 1)
          fcp[r] += __shfl_xor(fcp[r], m, 32);
      }
      if (lm == 0) {   // lanes 0 / 16: rows lh*8 .. lh*8+7
        float4 lo = make_float4(fcp[0], fcp[1], fcp[2], fcp[3]);
        float4 hi = make_float4(fcp[4], fcp[5], fcp[6], fcp[7]);
        float* pf = &s_fc[pp][wv * 16 + lh * 8];
        __builtin_memcpy(pf,     &lo, 16);   // ds_store_b128
        __builtin_memcpy(pf + 4, &hi, 16);
      }

      __syncthreads();   // the ONLY per-step barrier (ping-pong elsewhere)

      // reload full h_t as A-matrix (16-bit A layout): 4x ds_load_b128
      {
        const unsigned short* pr = &s_h[pp][lm * HDIM + lh * 16];
        __builtin_memcpy(&a0.f[0], pr,      16);   // K  0.. 7 (+16 for lh=1)
        __builtin_memcpy(&a0.f[1], pr + 8,  16);   // K  8..15
        __builtin_memcpy(&a1.f[0], pr + 32, 16);   // K 32..39
        __builtin_memcpy(&a1.f[1], pr + 40, 16);   // K 40..47
      }

      // wave0 lanes 0..15: combine 4 chunk partials per row, store output
      if (tid < 16) {
        const float* pf = s_fc[pp];
        float s = pf[tid] + pf[16 + tid] + pf[32 + tid] + pf[48 + tid];
        out[(base + tid) * TSEQ + t] = s + bfc;
      }

      pp ^= 1;
    }

    // chunk boundary: own async writes landed, then make visible to all
    wait_async_zero();
    __syncthreads();
    pb ^= 1;
  }
}

extern "C" void kernel_launch(void* const* d_in, const int* in_sizes, int n_in,
                              void* d_out, int out_size, void* d_ws, size_t ws_size,
                              hipStream_t stream) {
  const float* x    = (const float*)d_in[0];
  const float* w_ih = (const float*)d_in[1];
  const float* w_hh = (const float*)d_in[2];
  const float* b_ih = (const float*)d_in[3];
  const float* b_hh = (const float*)d_in[4];
  const float* w_fc = (const float*)d_in[5];
  const float* b_fc = (const float*)d_in[6];
  float* out = (float*)d_out;

  dim3 grid(BATCH / 16);   // 128 workgroups, 4 waves each
  dim3 block(128);
  hipLaunchKernelGGL(lstm_wmma_kernel, grid, block, 0, stream,
                     x, w_ih, w_hh, b_ih, b_hh, w_fc, b_fc, out);
}
